// SelfAttnE2VOpt_10290741641925
// MI455X (gfx1250) — compile-verified, matrix operand order checked by reading, not run on hardware
//
#include <hip/hip_runtime.h>
#include <hip/hip_bf16.h>
#include <math.h>

#define N_NODES 30000
#define N_EDGES 60000
#define K_ORD   6
#define N_TOK   (N_NODES + N_EDGES)            /* 90000 */
#define M_MSG   (N_EDGES * K_ORD + N_NODES)    /* 390000 */
#define D_MODEL 256
#define HYPER   256

typedef _Float16 f16_t;
typedef __attribute__((ext_vector_type(16))) _Float16 v16h;
typedef __attribute__((ext_vector_type(8)))  float    v8f;
typedef int v4i __attribute__((ext_vector_type(4)));

// ---------------------------------------------------------------------------
// gfx1250 async global->LDS staging (ASYNCcnt-tracked); guarded so a toolchain
// without the builtin still compiles (synchronous float4 fallback).
// Builtin prototype (from probe): (v4i as1*, v4i as3*, imm int off, imm int cpol)
// ---------------------------------------------------------------------------
#if defined(__AMDGCN__) && __has_builtin(__builtin_amdgcn_global_load_async_to_lds_b128)
#define ASYNC_LDS 1
#endif

__device__ __forceinline__ void async_b128(const void* g, void* l) {
#ifdef ASYNC_LDS
    __builtin_amdgcn_global_load_async_to_lds_b128(
        (__attribute__((address_space(1))) v4i*)(v4i*)g,
        (__attribute__((address_space(3))) v4i*)(v4i*)l,
        /*offset=*/0, /*cpol=*/0);
#else
    *(float4*)l = *(const float4*)g;
#endif
}

__device__ __forceinline__ void async_wait() {
#ifdef ASYNC_LDS
#if __has_builtin(__builtin_amdgcn_s_wait_asynccnt)
    __builtin_amdgcn_s_wait_asynccnt(0);
#else
    asm volatile("s_wait_asynccnt 0" ::: "memory");
#endif
#endif
}

// ---------------------------------------------------------------------------
// WMMA helpers (16x16x32 f16 -> f32)
// ---------------------------------------------------------------------------
__device__ __forceinline__ v8f wmma_f16(v16h a, v16h b, v8f c) {
    return __builtin_amdgcn_wmma_f32_16x16x32_f16(
        /*neg_a=*/false, a, /*neg_b=*/false, b,
        /*c_mod=*/(short)0, c, /*reuse_a=*/false, /*reuse_b=*/false);
}

// A fragment: 16x32 f16 tile, row-major source with leading dim `ld` (halves).
// Per ISA 7.12.2: lane L holds row M=L&15; lanes 0-15 cover K 0-7 & 16-23,
// lanes 16-31 cover K 8-15 & 24-31 (pairs per VGPR).
__device__ __forceinline__ v16h load_frag_A(const f16_t* base, int ld, int lane) {
    int row = lane & 15;
    int kb  = (lane >> 4) << 3;
    const f16_t* r = base + row * ld + kb;
    v16h a;
#pragma unroll
    for (int g = 0; g < 8; ++g) {
        int k = (g < 4) ? (2 * g) : (16 + 2 * (g - 4));
        a[2 * g]     = r[k];
        a[2 * g + 1] = r[k + 1];
    }
    return a;
}

// ---------------------------------------------------------------------------
// Weight packer: W[K][Nw] f32 row-major -> fragment-ordered f16 so that a
// wave's B fragment for (ktile,ntile) is 32B contiguous per lane.
// ---------------------------------------------------------------------------
__global__ void pack_weight_kernel(const float* __restrict__ W, f16_t* __restrict__ dst,
                                   int K, int Nw) {
    int total = K * Nw;
    for (int e = blockIdx.x * blockDim.x + threadIdx.x; e < total;
         e += gridDim.x * blockDim.x) {
        int hh   = e & 1;
        int g    = (e >> 1) & 7;
        int lane = (e >> 4) & 31;
        int t    = e >> 9;
        int ntiles = Nw >> 4;
        int nt = t % ntiles;
        int kt = t / ntiles;
        int kin = (g < 4) ? (2 * g + hh) : (16 + 2 * (g - 4) + hh);
        int k = kt * 32 + ((lane >> 4) << 3) + kin;
        int n = nt * 16 + (lane & 15);
        dst[e] = (f16_t)W[(size_t)k * Nw + n];
    }
}

// ---------------------------------------------------------------------------
// Workspace init (poisoned between runs -> must re-init every launch)
// ---------------------------------------------------------------------------
__global__ void init_ws_kernel(float* att1, float* segsum, float* segmax,
                               float* att0acc, float* gsum, float* gmax) {
    size_t i = (size_t)blockIdx.x * blockDim.x + threadIdx.x;
    size_t stride = (size_t)gridDim.x * blockDim.x;
    for (size_t j = i; j < (size_t)N_NODES * 256; j += stride) att1[j] = 0.f;
    for (size_t j = i; j < (size_t)N_NODES * 8; j += stride) {
        segsum[j] = 0.f;
        segmax[j] = -3.0e38f;
    }
    if (i < 256) att0acc[i] = 0.f;
    if (i < 8) { gsum[i] = 0.f; gmax[i] = -3.0e38f; }
}

// ---------------------------------------------------------------------------
// Small setup: PE tables, q-MLP, fold queries into Wq[256x16] and qb[16]
// ---------------------------------------------------------------------------
__global__ void small_setup_kernel(const float* __restrict__ q_w1, const float* __restrict__ q_b1,
                                   const float* __restrict__ q_w2, const float* __restrict__ q_b2,
                                   const float* __restrict__ k_w, const float* __restrict__ k_b,
                                   float* pe1, float* pe2, float* Wq, float* qbv) {
    __shared__ float peq[2][128];
    __shared__ float hq[2][256];
    __shared__ float qs[2][256];
    int tid = threadIdx.x;
    const float ln1e4 = 9.210340371976184f; // log(10000)

    for (int idx = tid; idx < 11 * 256; idx += 256) {
        int pos = idx >> 8, col = idx & 255;
        float div = expf(-(float)(col & ~1) * ln1e4 / 256.f);
        float ang = (float)pos * div;
        pe1[idx] = (col & 1) ? cosf(ang) : sinf(ang);
    }
    for (int idx = tid; idx < 2 * 256; idx += 256) {
        int pos = idx >> 8, col = idx & 255;
        float div = expf(-(float)(col & ~1) * ln1e4 / 256.f);
        float ang = (float)pos * div;
        pe2[idx] = (col & 1) ? cosf(ang) : sinf(ang);
    }
    for (int idx = tid; idx < 2 * 128; idx += 256) {
        int pos = idx >> 7, col = idx & 127;
        float div = expf(-(float)(col & ~1) * ln1e4 / 128.f);
        float ang = (float)pos * div;
        peq[pos][col] = (col & 1) ? cosf(ang) : sinf(ang);
    }
    __syncthreads();
    {   // q MLP layer 1: 128 -> 256, relu
        int j = tid;
        for (int pos = 0; pos < 2; ++pos) {
            float s = q_b1[j];
            for (int i = 0; i < 128; ++i) s += peq[pos][i] * q_w1[i * 256 + j];
            hq[pos][j] = s > 0.f ? s : 0.f;
        }
    }
    __syncthreads();
    {   // q MLP layer 2: 256 -> 256
        int d = tid;
        for (int pos = 0; pos < 2; ++pos) {
            float s = q_b2[d];
            for (int j = 0; j < 256; ++j) s += hq[pos][j] * q_w2[j * 256 + d];
            qs[pos][d] = s;
        }
    }
    __syncthreads();
    const float invs = 0.17677669529663687f; // 1/sqrt(32)
    {
        int d = tid;
        for (int c = 0; c < 16; ++c) {
            int h = c & 7, set = c >> 3;
            float s = 0.f;
            for (int dh = 0; dh < 32; ++dh)
                s += k_w[(size_t)d * 512 + set * 256 + h * 32 + dh] * qs[set][h * 32 + dh];
            Wq[d * 16 + c] = (set == 0) ? s * invs : s;
        }
    }
    if (tid < 16) {
        int h = tid & 7, set = tid >> 3;
        float s = 0.f;
        for (int dh = 0; dh < 32; ++dh)
            s += k_b[set * 256 + h * 32 + dh] * qs[set][h * 32 + dh];
        qbv[tid] = (set == 0) ? s * invs : s;
    }
}

// ---------------------------------------------------------------------------
// Fused token kernel: LN(n1)+MLP1 residual -> tok; logits (tok@Wq); V proj.
// 256 threads = 8 waves; each wave owns 2 column tiles of 16.
// ---------------------------------------------------------------------------
__global__ __launch_bounds__(256) void fused_token_kernel(
    const float* __restrict__ x_v, const float* __restrict__ x_e,
    const int* __restrict__ edge_orders,
    const float* __restrict__ n1_g, const float* __restrict__ n1_b,
    const float* __restrict__ pe1,
    const f16_t* __restrict__ m1w1p, const float* __restrict__ m1b1,
    const f16_t* __restrict__ m1w2p, const float* __restrict__ m1b2,
    const f16_t* __restrict__ vwp, const float* __restrict__ vb,
    const f16_t* __restrict__ Wqp, const float* __restrict__ qbv,
    f16_t* __restrict__ v_out, float* __restrict__ logit0, float* __restrict__ a1) {
    __shared__ float xrow[16][256];
    __shared__ f16_t lnpe[16][512];
    __shared__ f16_t h1s[16][256];
    __shared__ f16_t toks[16][256];
    __shared__ float red[16][16], red2[16][16];
    __shared__ float stat[16][2];

    int tid = threadIdx.x, lane = tid & 31, wave = tid >> 5;
    int row = tid >> 4, l16 = tid & 15;
    int t0 = blockIdx.x * 16;
    int tok = t0 + row;

    // --- Phase A: async-stage row into LDS, then LN(n1) + [ln | pe1[order]] ---
    const float* src;
    int order;
    if (tok < N_EDGES) { src = x_e + (size_t)tok * 256; order = edge_orders[tok]; }
    else               { src = x_v + (size_t)(tok - N_EDGES) * 256; order = 1; }
#pragma unroll
    for (int j = 0; j < 4; ++j) {
        int c = j * 16 + l16;                 // 64 x 16B chunks per row
        async_b128(src + c * 4, &xrow[row][c * 4]);
    }
    async_wait();
    float vals[16];
    float s = 0.f, s2 = 0.f;
#pragma unroll
    for (int j = 0; j < 16; ++j) {
        int d = j * 16 + l16;
        float x = xrow[row][d];
        vals[j] = x;
        s += x; s2 += x * x;
    }
    red[row][l16] = s; red2[row][l16] = s2;
    __syncthreads();
    if (l16 == 0) {
        float a = 0.f, b = 0.f;
        for (int i = 0; i < 16; ++i) { a += red[row][i]; b += red2[row][i]; }
        float mu = a * (1.f / 256.f);
        float var = b * (1.f / 256.f) - mu * mu;
        stat[row][0] = mu;
        stat[row][1] = rsqrtf(var + 1e-5f);
    }
    __syncthreads();
    float mu = stat[row][0], rs = stat[row][1];
#pragma unroll
    for (int j = 0; j < 16; ++j) {
        int d = j * 16 + l16;
        float ln = (vals[j] - mu) * rs * n1_g[d] + n1_b[d];
        lnpe[row][d]       = (f16_t)ln;
        lnpe[row][256 + d] = (f16_t)pe1[order * 256 + d];
    }
    __syncthreads();

    // --- Phase B: MLP1 layer1  [16x512]@[512x256] + b1, relu -> h1s ---
    for (int t = 0; t < 2; ++t) {
        int nt = wave + t * 8;
        v8f acc = {};
        for (int kt = 0; kt < 16; ++kt) {
            v16h a = load_frag_A(&lnpe[0][0] + kt * 32, 512, lane);
            const f16_t* bp = m1w1p + (((size_t)kt * 16 + nt) * 32 + lane) * 16;
            __builtin_prefetch(bp + 8192, 0, 1);
            v16h b = *(const v16h*)bp;
            acc = wmma_f16(a, b, acc);
        }
        int col = nt * 16 + (lane & 15);
        int rb = (lane >> 4) * 8;
        float bb = m1b1[col];
#pragma unroll
        for (int r = 0; r < 8; ++r) {
            float vv = acc[r] + bb;
            h1s[rb + r][col] = (f16_t)(vv > 0.f ? vv : 0.f);
        }
    }
    __syncthreads();

    // --- Phase C: MLP1 layer2 + residual -> toks (f16) ---
    for (int t = 0; t < 2; ++t) {
        int nt = wave + t * 8;
        v8f acc = {};
        for (int kt = 0; kt < 8; ++kt) {
            v16h a = load_frag_A(&h1s[0][0] + kt * 32, 256, lane);
            const f16_t* bp = m1w2p + (((size_t)kt * 16 + nt) * 32 + lane) * 16;
            v16h b = *(const v16h*)bp;
            acc = wmma_f16(a, b, acc);
        }
        int col = nt * 16 + (lane & 15);
        int rb = (lane >> 4) * 8;
        float bb = m1b2[col];
#pragma unroll
        for (int r = 0; r < 8; ++r)
            toks[rb + r][col] = (f16_t)(acc[r] + bb + xrow[rb + r][col]);
    }
    __syncthreads();

    // --- Phase D: logits tile (tok @ Wq[256x16]) ; wave 0 only ---
    if (wave == 0) {
        v8f acc = {};
        for (int kt = 0; kt < 8; ++kt) {
            v16h a = load_frag_A(&toks[0][0] + kt * 32, 256, lane);
            v16h b = *(const v16h*)(Wqp + ((size_t)kt * 32 + lane) * 16);
            acc = wmma_f16(a, b, acc);
        }
        int c = lane & 15;
        int rb = (lane >> 4) * 8;
#pragma unroll
        for (int r = 0; r < 8; ++r) {
            int tk = t0 + rb + r;
            float vv = acc[r] + qbv[c];
            if (c < 8) logit0[(size_t)tk * 8 + c] = vv;
            else       a1[(size_t)tk * 8 + (c - 8)] = vv;
        }
    }

    // --- Phase E: V projection (tok @ v_w + v_b) -> global f16 ---
    for (int t = 0; t < 2; ++t) {
        int nt = wave + t * 8;
        v8f acc = {};
        for (int kt = 0; kt < 8; ++kt) {
            v16h a = load_frag_A(&toks[0][0] + kt * 32, 256, lane);
            const f16_t* bp = vwp + (((size_t)kt * 16 + nt) * 32 + lane) * 16;
            __builtin_prefetch(bp + 8192, 0, 1);
            v16h b = *(const v16h*)bp;
            acc = wmma_f16(a, b, acc);
        }
        int col = nt * 16 + (lane & 15);
        int rb = (lane >> 4) * 8;
        float bb = vb[col];
#pragma unroll
        for (int r = 0; r < 8; ++r) {
            int tk = t0 + rb + r;
            v_out[(size_t)tk * 256 + col] = (f16_t)(acc[r] + bb);
        }
    }
}

// ---------------------------------------------------------------------------
// Float atomic max (sign-split int trick)
// ---------------------------------------------------------------------------
__device__ __forceinline__ void atomicMaxFloat(float* addr, float val) {
    if (val >= 0.f) atomicMax((int*)addr, __float_as_int(val));
    else            atomicMin((unsigned int*)addr, __float_as_uint(val));
}

// --- global PMA (att0): max, sum, weighted accumulate ---
__global__ void gmax_kernel(const float* __restrict__ logit0, float* gmax) {
    int u = blockIdx.x * blockDim.x + threadIdx.x;
    int total = gridDim.x * blockDim.x;
    int h = u & 7;
    float local = -3.0e38f;
    for (int row = u >> 3; row < N_TOK; row += (total >> 3))
        local = fmaxf(local, logit0[(size_t)row * 8 + h]);
    atomicMaxFloat(&gmax[h], local);
}

__global__ void gsum_kernel(const float* __restrict__ logit0, const float* __restrict__ gmax,
                            float* gsum) {
    int u = blockIdx.x * blockDim.x + threadIdx.x;
    int total = gridDim.x * blockDim.x;
    int h = u & 7;
    float gm = gmax[h];
    float local = 0.f;
    for (int row = u >> 3; row < N_TOK; row += (total >> 3))
        local += expf(logit0[(size_t)row * 8 + h] - gm);
    atomicAdd(&gsum[h], local);
}

__global__ void att0_acc_kernel(const float* __restrict__ logit0, const float* __restrict__ gmax,
                                const f16_t* __restrict__ v, float* att0acc) {
    int d = threadIdx.x;           // 0..255
    int h = d >> 5;
    float gm = gmax[h];
    float acc = 0.f;
    for (int tok = blockIdx.x; tok < N_TOK; tok += gridDim.x)
        acc += expf(logit0[(size_t)tok * 8 + h] - gm) * (float)v[(size_t)tok * 256 + d];
    atomicAdd(&att0acc[d], acc);
}

// att0 finalize: /gsum then LN(n2)+pe2[0] -> MLP2 residual (single row, fp32)
__global__ void att0_fin_kernel(const float* __restrict__ att0acc, const float* __restrict__ gsum,
                                const float* __restrict__ pe2,
                                const float* __restrict__ n2_g, const float* __restrict__ n2_b,
                                const float* __restrict__ m2_w1, const float* __restrict__ m2_b1,
                                const float* __restrict__ m2_w2, const float* __restrict__ m2_b2,
                                float* att0p) {
    __shared__ float inbuf[512];
    __shared__ float hbuf[256];
    __shared__ float red[256];
    __shared__ float stat2[2];
    int d = threadIdx.x;
    float a = att0acc[d] / gsum[d >> 5];
    red[d] = a;
    __syncthreads();
    if (d == 0) {
        float s = 0.f, s2 = 0.f;
        for (int i = 0; i < 256; ++i) { s += red[i]; s2 += red[i] * red[i]; }
        float mu = s * (1.f / 256.f);
        float var = s2 * (1.f / 256.f) - mu * mu;
        stat2[0] = mu; stat2[1] = rsqrtf(var + 1e-5f);
    }
    __syncthreads();
    inbuf[d]       = (a - stat2[0]) * stat2[1] * n2_g[d] + n2_b[d];
    inbuf[256 + d] = pe2[d];       // pe2 row 0
    __syncthreads();
    {
        int j = d;
        float sj = m2_b1[j];
        for (int i = 0; i < 512; ++i) sj += inbuf[i] * m2_w1[(size_t)i * 256 + j];
        hbuf[j] = sj > 0.f ? sj : 0.f;
    }
    __syncthreads();
    {
        float so = m2_b2[d];
        for (int j = 0; j < 256; ++j) so += hbuf[j] * m2_w2[(size_t)j * 256 + d];
        att0p[d] = a + so;
    }
}

// --- segment PMA (att1): leaky -> segment max -> exp/sum -> weighted scatter ---
__global__ void seg_max_kernel(const int* __restrict__ ei, const float* __restrict__ a1,
                               float* segmax) {
    int i = blockIdx.x * blockDim.x + threadIdx.x;
    if (i >= M_MSG * 8) return;
    int m = i >> 3, h = i & 7;
    int s = ei[m], t = ei[M_MSG + m];
    float la = a1[(size_t)s * 8 + h];
    la = la > 0.f ? la : 0.2f * la;
    atomicMaxFloat(&segmax[(size_t)t * 8 + h], la);
}

__global__ void seg_sum_kernel(const int* __restrict__ ei, const float* __restrict__ a1,
                               const float* __restrict__ segmax, float* emsg, float* segsum) {
    int i = blockIdx.x * blockDim.x + threadIdx.x;
    if (i >= M_MSG * 8) return;
    int m = i >> 3, h = i & 7;
    int s = ei[m], t = ei[M_MSG + m];
    float la = a1[(size_t)s * 8 + h];
    la = la > 0.f ? la : 0.2f * la;
    float e = expf(la - segmax[(size_t)t * 8 + h]);
    emsg[i] = e;
    atomicAdd(&segsum[(size_t)t * 8 + h], e);
}

// One 256-thread block per message: coalesced f16 gather of v[src] and
// coalesced f32 atomic scatter into att1[tgt].
__global__ __launch_bounds__(256) void seg_acc_kernel(
    const int* __restrict__ ei, const float* __restrict__ emsg,
    const float* __restrict__ segsum, const f16_t* __restrict__ v,
    float* __restrict__ att1) {
    int m = blockIdx.x;
    int d = threadIdx.x;
    int h = d >> 5;
    int s = ei[m], t = ei[M_MSG + m];
    float w = emsg[(size_t)m * 8 + h] / segsum[(size_t)t * 8 + h];
    atomicAdd(&att1[(size_t)t * 256 + d], w * (float)v[(size_t)s * 256 + d]);
}

// ---------------------------------------------------------------------------
// Node-final kernel: att1' = att1 + MLP2([LN_n2(att1)|pe2[1]]); x = att0'+att1'
//                    out   = x + MLP3(LN_n3(x)) + bias_v
// ---------------------------------------------------------------------------
__global__ __launch_bounds__(256) void node_final_kernel(
    const float* __restrict__ att1, const float* __restrict__ att0p,
    const float* __restrict__ pe2,
    const float* __restrict__ n2_g, const float* __restrict__ n2_b,
    const float* __restrict__ n3_g, const float* __restrict__ n3_b,
    const f16_t* __restrict__ m2w1p, const float* __restrict__ m2b1,
    const f16_t* __restrict__ m2w2p, const float* __restrict__ m2b2,
    const f16_t* __restrict__ m3w1p, const float* __restrict__ m3b1,
    const f16_t* __restrict__ m3w2p, const float* __restrict__ m3b2,
    const float* __restrict__ bias_v, float* __restrict__ out) {
    __shared__ float xrow[16][256];
    __shared__ f16_t lnpe[16][512];
    __shared__ f16_t h1s[16][256];
    __shared__ float red[16][16], red2[16][16];
    __shared__ float stat[16][2];

    int tid = threadIdx.x, lane = tid & 31, wave = tid >> 5;
    int row = tid >> 4, l16 = tid & 15;
    int n0 = blockIdx.x * 16;

    // --- Phase A: async-stage att1 row, LN(n2), [ln | pe2[1]] ---
    const float* src = att1 + (size_t)(n0 + row) * 256;
#pragma unroll
    for (int j = 0; j < 4; ++j) {
        int c = j * 16 + l16;
        async_b128(src + c * 4, &xrow[row][c * 4]);
    }
    async_wait();
    float vals[16];
    float s = 0.f, s2 = 0.f;
#pragma unroll
    for (int j = 0; j < 16; ++j) {
        int d = j * 16 + l16;
        float x = xrow[row][d];
        vals[j] = x;
        s += x; s2 += x * x;
    }
    red[row][l16] = s; red2[row][l16] = s2;
    __syncthreads();
    if (l16 == 0) {
        float a = 0.f, b = 0.f;
        for (int i = 0; i < 16; ++i) { a += red[row][i]; b += red2[row][i]; }
        float mu = a * (1.f / 256.f);
        float var = b * (1.f / 256.f) - mu * mu;
        stat[row][0] = mu; stat[row][1] = rsqrtf(var + 1e-5f);
    }
    __syncthreads();
    {
        float mu = stat[row][0], rs = stat[row][1];
#pragma unroll
        for (int j = 0; j < 16; ++j) {
            int d = j * 16 + l16;
            lnpe[row][d]       = (f16_t)((vals[j] - mu) * rs * n2_g[d] + n2_b[d]);
            lnpe[row][256 + d] = (f16_t)pe2[256 + d];   // pe2 row 1
        }
    }
    __syncthreads();

    // --- Phase B: MLP2 layer1 (512->256 relu) ---
    for (int t = 0; t < 2; ++t) {
        int nt = wave + t * 8;
        v8f acc = {};
        for (int kt = 0; kt < 16; ++kt) {
            v16h a = load_frag_A(&lnpe[0][0] + kt * 32, 512, lane);
            const f16_t* bp = m2w1p + (((size_t)kt * 16 + nt) * 32 + lane) * 16;
            __builtin_prefetch(bp + 8192, 0, 1);
            v16h b = *(const v16h*)bp;
            acc = wmma_f16(a, b, acc);
        }
        int col = nt * 16 + (lane & 15);
        int rb = (lane >> 4) * 8;
        float bb = m2b1[col];
#pragma unroll
        for (int r = 0; r < 8; ++r) {
            float vv = acc[r] + bb;
            h1s[rb + r][col] = (f16_t)(vv > 0.f ? vv : 0.f);
        }
    }
    __syncthreads();

    // --- Phase C: MLP2 layer2, residual, + att0' -> x in xrow ---
    for (int t = 0; t < 2; ++t) {
        int nt = wave + t * 8;
        v8f acc = {};
        for (int kt = 0; kt < 8; ++kt) {
            v16h a = load_frag_A(&h1s[0][0] + kt * 32, 256, lane);
            v16h b = *(const v16h*)(m2w2p + (((size_t)kt * 16 + nt) * 32 + lane) * 16);
            acc = wmma_f16(a, b, acc);
        }
        int col = nt * 16 + (lane & 15);
        int rb = (lane >> 4) * 8;
        float bb = m2b2[col] + att0p[col];
#pragma unroll
        for (int r = 0; r < 8; ++r)
            xrow[rb + r][col] = xrow[rb + r][col] + acc[r] + bb;
    }
    __syncthreads();

    // --- Phase D: LN(n3) of x ---
    s = 0.f; s2 = 0.f;
#pragma unroll
    for (int j = 0; j < 16; ++j) {
        int d = j * 16 + l16;
        float x = xrow[row][d];
        vals[j] = x;
        s += x; s2 += x * x;
    }
    red[row][l16] = s; red2[row][l16] = s2;
    __syncthreads();
    if (l16 == 0) {
        float a = 0.f, b = 0.f;
        for (int i = 0; i < 16; ++i) { a += red[row][i]; b += red2[row][i]; }
        float mu = a * (1.f / 256.f);
        float var = b * (1.f / 256.f) - mu * mu;
        stat[row][0] = mu; stat[row][1] = rsqrtf(var + 1e-5f);
    }
    __syncthreads();
    {
        float mu = stat[row][0], rs = stat[row][1];
#pragma unroll
        for (int j = 0; j < 16; ++j) {
            int d = j * 16 + l16;
            lnpe[row][d] = (f16_t)((vals[j] - mu) * rs * n3_g[d] + n3_b[d]);
        }
    }
    __syncthreads();

    // --- Phase E: MLP3 layer1 (256->256 relu) ---
    for (int t = 0; t < 2; ++t) {
        int nt = wave + t * 8;
        v8f acc = {};
        for (int kt = 0; kt < 8; ++kt) {
            v16h a = load_frag_A(&lnpe[0][0] + kt * 32, 512, lane);
            v16h b = *(const v16h*)(m3w1p + (((size_t)kt * 16 + nt) * 32 + lane) * 16);
            acc = wmma_f16(a, b, acc);
        }
        int col = nt * 16 + (lane & 15);
        int rb = (lane >> 4) * 8;
        float bb = m3b1[col];
#pragma unroll
        for (int r = 0; r < 8; ++r) {
            float vv = acc[r] + bb;
            h1s[rb + r][col] = (f16_t)(vv > 0.f ? vv : 0.f);
        }
    }
    __syncthreads();

    // --- Phase F: MLP3 layer2 + residual + bias_v -> out ---
    for (int t = 0; t < 2; ++t) {
        int nt = wave + t * 8;
        v8f acc = {};
        for (int kt = 0; kt < 8; ++kt) {
            v16h a = load_frag_A(&h1s[0][0] + kt * 32, 256, lane);
            v16h b = *(const v16h*)(m3w2p + (((size_t)kt * 16 + nt) * 32 + lane) * 16);
            acc = wmma_f16(a, b, acc);
        }
        int col = nt * 16 + (lane & 15);
        int rb = (lane >> 4) * 8;
        float bb = m3b2[col] + bias_v[col];
#pragma unroll
        for (int r = 0; r < 8; ++r) {
            int node = n0 + rb + r;
            out[(size_t)node * 256 + col] = xrow[rb + r][col] + acc[r] + bb;
        }
    }
}

// ---------------------------------------------------------------------------
// Launch
// ---------------------------------------------------------------------------
extern "C" void kernel_launch(void* const* d_in, const int* in_sizes, int n_in,
                              void* d_out, int out_size, void* d_ws, size_t ws_size,
                              hipStream_t stream) {
    const float* x_v  = (const float*)d_in[0];
    const float* x_e  = (const float*)d_in[1];
    const int* edge_orders = (const int*)d_in[2];
    const int* edge_index  = (const int*)d_in[3];
    const float* q_w1 = (const float*)d_in[4];
    const float* q_b1 = (const float*)d_in[5];
    const float* q_w2 = (const float*)d_in[6];
    const float* q_b2 = (const float*)d_in[7];
    const float* k_w  = (const float*)d_in[8];
    const float* k_b  = (const float*)d_in[9];
    const float* v_w  = (const float*)d_in[10];
    const float* v_b  = (const float*)d_in[11];
    const float* m1_w1 = (const float*)d_in[12];
    const float* m1_b1 = (const float*)d_in[13];
    const float* m1_w2 = (const float*)d_in[14];
    const float* m1_b2 = (const float*)d_in[15];
    const float* m2_w1 = (const float*)d_in[16];
    const float* m2_b1 = (const float*)d_in[17];
    const float* m2_w2 = (const float*)d_in[18];
    const float* m2_b2 = (const float*)d_in[19];
    const float* m3_w1 = (const float*)d_in[20];
    const float* m3_b1 = (const float*)d_in[21];
    const float* m3_w2 = (const float*)d_in[22];
    const float* m3_b2 = (const float*)d_in[23];
    const float* n1_g = (const float*)d_in[24];
    const float* n1_b = (const float*)d_in[25];
    const float* n2_g = (const float*)d_in[26];
    const float* n2_b = (const float*)d_in[27];
    const float* n3_g = (const float*)d_in[28];
    const float* n3_b = (const float*)d_in[29];
    const float* bias_v = (const float*)d_in[30];
    float* out = (float*)d_out;

    // ---- workspace carve-out (256B aligned slabs) ----
    char* p = (char*)d_ws;
    auto alloc = [&](size_t bytes) -> void* {
        void* r = (void*)p;
        p += (bytes + 255) & ~(size_t)255;
        return r;
    };
    float* pe1     = (float*)alloc(11 * 256 * 4);
    float* pe2     = (float*)alloc(2 * 256 * 4);
    float* Wq      = (float*)alloc(256 * 16 * 4);
    float* qbv     = (float*)alloc(16 * 4);
    float* logit0  = (float*)alloc((size_t)N_TOK * 8 * 4);
    float* a1      = (float*)alloc((size_t)N_TOK * 8 * 4);
    f16_t* vbuf    = (f16_t*)alloc((size_t)N_TOK * 256 * 2);
    float* segmax  = (float*)alloc((size_t)N_NODES * 8 * 4);
    float* segsum  = (float*)alloc((size_t)N_NODES * 8 * 4);
    float* emsg    = (float*)alloc((size_t)M_MSG * 8 * 4);
    float* att1    = (float*)alloc((size_t)N_NODES * 256 * 4);
    float* att0acc = (float*)alloc(256 * 4);
    float* gsum    = (float*)alloc(8 * 4);
    float* gmax    = (float*)alloc(8 * 4);
    float* att0p   = (float*)alloc(256 * 4);
    f16_t* m1w1p = (f16_t*)alloc(512 * 256 * 2);
    f16_t* m1w2p = (f16_t*)alloc(256 * 256 * 2);
    f16_t* vwp   = (f16_t*)alloc(256 * 256 * 2);
    f16_t* m2w1p = (f16_t*)alloc(512 * 256 * 2);
    f16_t* m2w2p = (f16_t*)alloc(256 * 256 * 2);
    f16_t* m3w1p = (f16_t*)alloc(256 * 256 * 2);
    f16_t* m3w2p = (f16_t*)alloc(256 * 256 * 2);
    f16_t* Wqp   = (f16_t*)alloc(256 * 16 * 2);

    // 1) init accumulators
    init_ws_kernel<<<2048, 256, 0, stream>>>(att1, segsum, segmax, att0acc, gsum, gmax);
    // 2) PE tables + q MLP + folded query weights
    small_setup_kernel<<<1, 256, 0, stream>>>(q_w1, q_b1, q_w2, q_b2, k_w, k_b,
                                              pe1, pe2, Wq, qbv);
    // 3) pack weights to WMMA fragment order (f16)
    pack_weight_kernel<<<512, 256, 0, stream>>>(m1_w1, m1w1p, 512, 256);
    pack_weight_kernel<<<256, 256, 0, stream>>>(m1_w2, m1w2p, 256, 256);
    pack_weight_kernel<<<256, 256, 0, stream>>>(v_w, vwp, 256, 256);
    pack_weight_kernel<<<512, 256, 0, stream>>>(m2_w1, m2w1p, 512, 256);
    pack_weight_kernel<<<256, 256, 0, stream>>>(m2_w2, m2w2p, 256, 256);
    pack_weight_kernel<<<256, 256, 0, stream>>>(m3_w1, m3w1p, 256, 256);
    pack_weight_kernel<<<256, 256, 0, stream>>>(m3_w2, m3w2p, 256, 256);
    pack_weight_kernel<<<16, 256, 0, stream>>>(Wq, Wqp, 256, 16);
    // 4) fused token pass (LN+MLP1+logits+V), 90000/16 tiles
    fused_token_kernel<<<N_TOK / 16, 256, 0, stream>>>(
        x_v, x_e, edge_orders, n1_g, n1_b, pe1,
        m1w1p, m1_b1, m1w2p, m1_b2, vwp, v_b, Wqp, qbv,
        vbuf, logit0, a1);
    // 5) global PMA -> att0'
    gmax_kernel<<<64, 256, 0, stream>>>(logit0, gmax);
    gsum_kernel<<<64, 256, 0, stream>>>(logit0, gmax, gsum);
    att0_acc_kernel<<<128, 256, 0, stream>>>(logit0, gmax, vbuf, att0acc);
    att0_fin_kernel<<<1, 256, 0, stream>>>(att0acc, gsum, pe2, n2_g, n2_b,
                                           m2_w1, m2_b1, m2_w2, m2_b2, att0p);
    // 6) segment PMA -> att1
    int segBlocks = (M_MSG * 8 + 255) / 256;
    seg_max_kernel<<<segBlocks, 256, 0, stream>>>(edge_index, a1, segmax);
    seg_sum_kernel<<<segBlocks, 256, 0, stream>>>(edge_index, a1, segmax, emsg, segsum);
    seg_acc_kernel<<<M_MSG, 256, 0, stream>>>(edge_index, emsg, segsum, vbuf, att1);
    // 7) node finalize (MLP2 residual + att0 + MLP3 + bias)
    node_final_kernel<<<N_NODES / 16, 256, 0, stream>>>(
        att1, att0p, pe2, n2_g, n2_b, n3_g, n3_b,
        m2w1p, m2_b1, m2w2p, m2_b2, m3w1p, m3_b1, m3w2p, m3_b2,
        bias_v, out);
}
